// MultiHeadedLocationAwareAttention_38199439131211
// MI455X (gfx1250) — matrix operand order repaired and checked by reading, not compiled, not run on hardware
//
#include <hip/hip_runtime.h>

// ---------------- problem constants ----------------
constexpr int B = 16, S = 8192, F = 1024, H = 8, C = 10, D = 128;
constexpr int TM = 32;              // s-rows per block in the fused score kernel
constexpr int APAD = 8;             // bf16 row padding (16B) -> conflict-free ds reads
constexpr int ALD  = F + APAD;      // LDS A row stride in bf16 elements

// ---------------- CDNA5 WMMA / vector types ----------------
typedef __attribute__((ext_vector_type(16))) __bf16 v16bf;
typedef __attribute__((ext_vector_type(8)))  __bf16 v8bf;
typedef __attribute__((ext_vector_type(8)))  float  v8f;
typedef __attribute__((ext_vector_type(4)))  float  v4f;   // native vec4 for NT loads

struct __align__(8) bf4 { __bf16 x, y, z, w; };

__device__ __forceinline__ v16bf cat8(v8bf lo, v8bf hi) {
  return __builtin_shufflevector(lo, hi, 0,1,2,3,4,5,6,7,8,9,10,11,12,13,14,15);
}

// ---------------- kernel 1: v_w f32 -> bf16 ----------------
__global__ __launch_bounds__(256) void cvt_kernel(const float* __restrict__ w,
                                                  __bf16* __restrict__ o) {
  int i = (blockIdx.x * 256 + threadIdx.x) * 4;
  v4f v = *(const v4f*)(w + i);
  bf4 r{(__bf16)v.x, (__bf16)v.y, (__bf16)v.z, (__bf16)v.w};
  *(bf4*)(o + i) = r;
}

// ---------------- kernel 2: q projection (tiny GEMV) ----------------
__global__ __launch_bounds__(256) void qproj_kernel(const float* __restrict__ query,
                                                    const float* __restrict__ q_w,
                                                    float* __restrict__ qp) {
  int gid = blockIdx.x * 256 + threadIdx.x;
  int o = gid >> 4, l = gid & 15;          // 16 lanes per output
  int b = o / F, f = o % F;
  const float* qr = query + (size_t)b * F;
  const float* wr = q_w + (size_t)f * F;
  float acc = 0.f;
  for (int k = l; k < F; k += 16) acc += qr[k] * wr[k];
  #pragma unroll
  for (int m = 1; m < 16; m <<= 1) acc += __shfl_xor(acc, m, 32);
  if (l == 0) qp[o] = acc;
}

// ---------------- kernel 3: fused conv + loc_energy + bf16 WMMA projection
//                  + additive energy + tanh + score reduction ----------------
__global__ __launch_bounds__(256) void score_kernel(
    const float*  __restrict__ value,      // [B,S,F]  (streamed, NT)
    const __bf16* __restrict__ vw,         // [F,F] bf16 (row n = out feature, k contiguous)
    const float*  __restrict__ qp,         // [B,F]
    const float*  __restrict__ prev_align, // [B,H,S]
    const float*  __restrict__ conv_w,     // [C,H,3]
    const float*  __restrict__ conv_b,     // [C]
    const float*  __restrict__ loc_w,      // [D,C]
    const float*  __restrict__ bias,       // [D]
    const float*  __restrict__ score_w,    // [D]
    const float*  __restrict__ score_b,    // [1]
    float*        __restrict__ score)      // [B,H,S]
{
  __shared__ __bf16 Alds[TM * ALD];        // 66.0 KB  value tile (bf16)
  __shared__ float  locE[TM][D];           // 16.0 KB  tanh(loc @ loc_w)
  __shared__ float  locc[C][TM];           //  1.3 KB  conv output
  __shared__ float  qs[F];                 //  4.0 KB  projected q (all heads)
  __shared__ float  locw_s[D * C];         //  5.0 KB
  __shared__ float  sws[D];
  __shared__ float  biass[D];

  const int b    = blockIdx.y;
  const int s0   = blockIdx.x * TM;
  const int t    = threadIdx.x;
  const int wave = t >> 5;                 // wave32: wave == head
  const int lane = t & 31;
  const int lq   = lane & 15;
  const int lh   = lane >> 4;

  // ---- stage value tile -> bf16 LDS (coalesced NT loads, 8B LDS stores) ----
  const float* vrow = value + ((size_t)b * S + s0) * F;
  for (int i = t; i < TM * (F / 4); i += 256) {
    int r = i >> 8, c4 = i & 255;
    v4f v4 = __builtin_nontemporal_load(((const v4f*)(vrow + (size_t)r * F)) + c4);
    bf4 pk{(__bf16)v4.x, (__bf16)v4.y, (__bf16)v4.z, (__bf16)v4.w};
    *(bf4*)&Alds[r * ALD + c4 * 4] = pk;
  }
  for (int i = t; i < F; i += 256) qs[i] = qp[(size_t)b * F + i];
  for (int i = t; i < D * C; i += 256) locw_s[i] = loc_w[i];
  if (t < D) { sws[t] = score_w[t]; biass[t] = bias[t]; }

  // ---- conv1d (SAME, K=3) over prev_align: [H,S] -> [C, TM] ----
  for (int i = t; i < C * TM; i += 256) {
    int c = i / TM, r = i % TM, s = s0 + r;
    float acc = conv_b[c];
    #pragma unroll
    for (int h = 0; h < H; ++h) {
      const float* pa = prev_align + ((size_t)b * H + h) * S;
      const float* w3 = conv_w + (c * H + h) * 3;
      float x0 = (s > 0)     ? pa[s - 1] : 0.f;
      float x2 = (s < S - 1) ? pa[s + 1] : 0.f;
      acc += x0 * w3[0] + pa[s] * w3[1] + x2 * w3[2];
    }
    locc[c][r] = acc;
  }
  __syncthreads();

  // ---- loc_energy = tanh(loc @ loc_w) : [TM, D] ----
  for (int i = t; i < TM * D; i += 256) {
    int r = i / D, d = i % D;
    float acc = 0.f;
    #pragma unroll
    for (int c = 0; c < C; ++c) acc += locc[c][r] * locw_s[d * C + c];
    locE[r][d] = tanhf(acc);
  }
  __syncthreads();

  // ---- per-wave GEMM over its head's 128 output columns ----
  // A layout (matches B): lane<16 holds K {k0..k0+15}; lane>=16 holds K {k0+16..k0+31}.
  const int h = wave;
  const int kA = lh * 16;                  // A-fragment K base within 32-chunk
  const int kB = lh * 16;                  // B-fragment K base within 32-chunk
  const __bf16* a0base = &Alds[lq * ALD + kA];         // rows 0..15
  const __bf16* a1base = &Alds[(lq + 16) * ALD + kA];  // rows 16..31

  float part0[8], part1[8];
  #pragma unroll
  for (int r = 0; r < 8; ++r) { part0[r] = 0.f; part1[r] = 0.f; }

  for (int nt = 0; nt < 8; ++nt) {
    const int d = nt * 16 + lq;            // column within head (also lane's D index)
    const __bf16* bptr = vw + ((size_t)(h * 128 + d)) * F + kB;  // 32 contiguous B per step
    v8f acc0 = {}, acc1 = {};
    #pragma unroll 4
    for (int k0 = 0; k0 < F; k0 += 32) {
      __builtin_prefetch(bptr + k0 + 128, 0, 3);       // WGP-scope global_prefetch_b8
      v16bf bfrag  = *(const v16bf*)(bptr + k0);
      v16bf afrag0 = cat8(*(const v8bf*)(a0base + k0), *(const v8bf*)(a0base + k0 + 8));
      v16bf afrag1 = cat8(*(const v8bf*)(a1base + k0), *(const v8bf*)(a1base + k0 + 8));
      acc0 = __builtin_amdgcn_wmma_f32_16x16x32_bf16(false, afrag0, false, bfrag,
                                                     (short)0, acc0, false, false);
      acc1 = __builtin_amdgcn_wmma_f32_16x16x32_bf16(false, afrag1, false, bfrag,
                                                     (short)0, acc1, false, false);
    }
    // fused epilogue: energy = tanh(v + q + locE + bias); partial += energy * score_w[d]
    const float qv = qs[h * 128 + d];
    const float sw = sws[d];
    const float bv = biass[d];
    #pragma unroll
    for (int r = 0; r < 8; ++r) {
      int m0 = r + lh * 8;                 // D-matrix M = r + 8*(lane>=16)
      part0[r] += tanhf(acc0[r] + qv + locE[m0][d] + bv) * sw;
      part1[r] += tanhf(acc1[r] + qv + locE[16 + m0][d] + bv) * sw;
    }
  }

  // ---- reduce the 16 D-lanes sharing each row, write scores ----
  #pragma unroll
  for (int r = 0; r < 8; ++r) {
    float p0 = part0[r], p1 = part1[r];
    #pragma unroll
    for (int m = 1; m < 16; m <<= 1) {
      p0 += __shfl_xor(p0, m, 32);
      p1 += __shfl_xor(p1, m, 32);
    }
    part0[r] = p0; part1[r] = p1;
  }
  if (lq == 0) {
    const float sb = score_b[0];
    float* srow = score + ((size_t)b * H + h) * S + s0;
    #pragma unroll
    for (int r = 0; r < 8; ++r) {
      srow[r + lh * 8]      = part0[r] + sb;   // tile rows 0..15
      srow[16 + r + lh * 8] = part1[r] + sb;   // tile rows 16..31
    }
  }
}

// ---------------- kernel 4: softmax over S per (b,h) ----------------
__global__ __launch_bounds__(256) void softmax_kernel(const float* __restrict__ score,
                                                      float* __restrict__ align) {
  const int bh = blockIdx.x;
  const float* src = score + (size_t)bh * S;
  float* dst = align + (size_t)bh * S;
  const int t = threadIdx.x;
  __shared__ float smax[8], ssum[8];

  float vals[S / 256];
  float mx = -INFINITY;
  #pragma unroll
  for (int i = 0; i < S / 256; ++i) { vals[i] = src[t + i * 256]; mx = fmaxf(mx, vals[i]); }
  #pragma unroll
  for (int m = 16; m; m >>= 1) mx = fmaxf(mx, __shfl_xor(mx, m, 32));
  if ((t & 31) == 0) smax[t >> 5] = mx;
  __syncthreads();
  float bm = smax[0];
  #pragma unroll
  for (int i = 1; i < 8; ++i) bm = fmaxf(bm, smax[i]);

  float sum = 0.f;
  #pragma unroll
  for (int i = 0; i < S / 256; ++i) { vals[i] = __expf(vals[i] - bm); sum += vals[i]; }
  #pragma unroll
  for (int m = 16; m; m >>= 1) sum += __shfl_xor(sum, m, 32);
  if ((t & 31) == 0) ssum[t >> 5] = sum;
  __syncthreads();
  float tot = 0.f;
  #pragma unroll
  for (int i = 0; i < 8; ++i) tot += ssum[i];
  const float inv = 1.0f / tot;
  #pragma unroll
  for (int i = 0; i < S / 256; ++i) dst[t + i * 256] = vals[i] * inv;
}

// ---------------- kernel 5: context = align . value(raw heads) ----------------
__global__ __launch_bounds__(256) void context_kernel(const float* __restrict__ align,
                                                      const float* __restrict__ value,
                                                      float* __restrict__ context) {
  const int bh = blockIdx.x;               // 0..127
  const int chunk = blockIdx.y;            // 0..7 slice of S
  const int b = bh >> 3, h = bh & 7;
  const int t = threadIdx.x;
  const int d = t & 127, half = t >> 7;
  const float* al = align + (size_t)bh * S;
  const float* vp = value + (size_t)b * S * F + h * 128 + d;
  const int sEnd = (chunk + 1) * (S / 8);
  float acc = 0.f;
  for (int s = chunk * (S / 8) + half; s < sEnd; s += 2)
    acc += al[s] * __builtin_nontemporal_load(vp + (size_t)s * F);
  __shared__ float tmp[256];
  tmp[t] = acc;
  __syncthreads();
  if (half == 0)
    atomicAdd(&context[((size_t)b * H + h) * D + d], tmp[t] + tmp[t + 128]);
}

// ---------------- kernel 6: output = [context||query] @ out_w.T + out_b ----------------
__global__ __launch_bounds__(256) void out_kernel(const float* __restrict__ context,
                                                  const float* __restrict__ query,
                                                  const float* __restrict__ out_w,
                                                  const float* __restrict__ out_b,
                                                  float* __restrict__ out) {
  int gid = blockIdx.x * 256 + threadIdx.x;
  int o = gid >> 4, l = gid & 15;
  int b = o / F, f = o % F;
  const float* wr = out_w + (size_t)f * (2 * F);
  const float* cb = context + (size_t)b * F;   // [B, H*D] flat
  const float* qb = query + (size_t)b * F;
  float acc = 0.f;
  for (int k = l; k < F; k += 16) acc += cb[k] * wr[k] + qb[k] * wr[F + k];
  #pragma unroll
  for (int m = 1; m < 16; m <<= 1) acc += __shfl_xor(acc, m, 32);
  if (l == 0) out[o] = acc + out_b[f];
}

// ---------------- host launch ----------------
extern "C" void kernel_launch(void* const* d_in, const int* in_sizes, int n_in,
                              void* d_out, int out_size, void* d_ws, size_t ws_size,
                              hipStream_t stream) {
  const float* query      = (const float*)d_in[0];
  const float* value      = (const float*)d_in[1];
  const float* prev_align = (const float*)d_in[2];
  const float* conv_w     = (const float*)d_in[3];
  const float* conv_b     = (const float*)d_in[4];
  const float* loc_w      = (const float*)d_in[5];
  const float* q_w        = (const float*)d_in[6];
  const float* v_w        = (const float*)d_in[7];
  const float* bias       = (const float*)d_in[8];
  const float* score_w    = (const float*)d_in[9];
  const float* score_b    = (const float*)d_in[10];
  const float* out_w      = (const float*)d_in[11];
  const float* out_b      = (const float*)d_in[12];

  char* ws = (char*)d_ws;
  size_t off = 0;
  auto carve = [&](size_t bytes) -> void* {
    void* p = ws + off;
    off += (bytes + 255) & ~(size_t)255;
    return p;
  };
  __bf16* vw_bf   = (__bf16*)carve((size_t)F * F * sizeof(__bf16));
  float*  score   = (float*)carve((size_t)B * H * S * sizeof(float));
  float*  qp      = (float*)carve((size_t)B * F * sizeof(float));
  float*  context = (float*)carve((size_t)B * F * sizeof(float));

  float* out_main  = (float*)d_out;        // [B,1,F]
  float* out_align = out_main + B * F;     // [B,H,S]

  cvt_kernel<<<(F * F) / (256 * 4), 256, 0, stream>>>(v_w, vw_bf);
  qproj_kernel<<<(B * F * 16) / 256, 256, 0, stream>>>(query, q_w, qp);
  (void)hipMemsetAsync(context, 0, (size_t)B * F * sizeof(float), stream);

  dim3 gScore(S / TM, B);
  score_kernel<<<gScore, 256, 0, stream>>>(value, vw_bf, qp, prev_align, conv_w,
                                           conv_b, loc_w, bias, score_w, score_b,
                                           score);
  softmax_kernel<<<B * H, 256, 0, stream>>>(score, out_align);
  dim3 gCtx(B * H, 8);
  context_kernel<<<gCtx, 256, 0, stream>>>(out_align, value, context);
  out_kernel<<<(B * F * 16) / 256, 256, 0, stream>>>(context, query, out_w, out_b,
                                                     out_main);
}
